// MultiScaleVectorQuantizer_29102698397969
// MI455X (gfx1250) — matrix operand order repaired and checked by reading, not compiled, not run on hardware
//
#include <hip/hip_runtime.h>
#include <hip/hip_bf16.h>
#include <math.h>

typedef __attribute__((ext_vector_type(16))) _Float16 v16h;
typedef __attribute__((ext_vector_type(8)))  _Float16 v8h;
typedef __attribute__((ext_vector_type(8)))  float    v8f;

#define C_DIM   64
#define HW      32
#define BATCH   32
#define NPIX    (BATCH*HW*HW)        // 32768 pixels
#define NTOT    (NPIX*C_DIM)         // 2097152 elements
#define VCODES  8192
#define NSCALES 6
#define KCONV   (C_DIM*9)            // 576
#define COMMIT_C 0.25f
#define RATIO_C  0.5f

// ---------------------------------------------------------------------------
// init: residual = f, recon = 0, loss accumulators = 0
// ---------------------------------------------------------------------------
__global__ void init_kernel(const float* __restrict__ f, float* __restrict__ residual,
                            float* __restrict__ recon, float* __restrict__ lossAcc) {
  int i = blockIdx.x * blockDim.x + threadIdx.x;
  if (i < NTOT) { residual[i] = f[i]; recon[i] = 0.0f; }
  if (i < NSCALES) lossAcc[i] = 0.0f;
}

// ---------------------------------------------------------------------------
// codebook squared norms + hi/lo f16 split (done ONCE, reused by all scales)
// ---------------------------------------------------------------------------
__global__ void cbsq_kernel(const float* __restrict__ cb, float* __restrict__ cbsq) {
  int v = blockIdx.x * blockDim.x + threadIdx.x;
  if (v >= VCODES) return;
  const float* row = cb + (size_t)v * C_DIM;
  float s = 0.0f;
  #pragma unroll 8
  for (int k = 0; k < C_DIM; ++k) s += row[k] * row[k];
  cbsq[v] = s;
}

__global__ void cvt_cb_kernel(const float* __restrict__ cb,
                              _Float16* __restrict__ cbh, _Float16* __restrict__ cbl) {
  int i = blockIdx.x * blockDim.x + threadIdx.x;
  if (i >= VCODES * C_DIM) return;
  float x = cb[i];
  _Float16 h = (_Float16)x;
  cbh[i] = h;
  cbl[i] = (_Float16)(x - (float)h);
}

// ---------------------------------------------------------------------------
// repack conv weights to f16: wrh[o][tap*64 + c] = phi_w[scale][o][c][ky][kx]
// ---------------------------------------------------------------------------
__global__ void repack_w_kernel(const float* __restrict__ phi_w, _Float16* __restrict__ wrh,
                                int scale) {
  int i = blockIdx.x * blockDim.x + threadIdx.x;
  if (i >= C_DIM * KCONV) return;
  int o = i / KCONV, K = i % KCONV;
  int tap = K >> 6, c = K & 63;
  wrh[i] = (_Float16)phi_w[(((size_t)scale * C_DIM + o) * C_DIM + c) * 9 + tap];
}

// ---------------------------------------------------------------------------
// area downsample residual (NCHW) -> cur (NHWC rows [32*r*r, 64])
// ---------------------------------------------------------------------------
__global__ void down_kernel(const float* __restrict__ residual, float* __restrict__ cur, int r) {
  int total = BATCH * r * r * C_DIM;
  int tid = blockIdx.x * blockDim.x + threadIdx.x;
  if (tid >= total) return;
  int c = tid & 63;
  int rest = tid >> 6;
  int x = rest % r; rest /= r;
  int y = rest % r;
  int b = rest / r;
  int k = HW / r;
  const float* base = residual + (((size_t)(b * C_DIM + c) * HW + y * k) * HW + x * k);
  float s = 0.0f;
  for (int dy = 0; dy < k; ++dy)
    for (int dx = 0; dx < k; ++dx)
      s += base[dy * HW + dx];
  cur[tid] = s / (float)(k * k);
}

// ---------------------------------------------------------------------------
// WMMA nearest-neighbor search. One wave handles 16 rows vs all 8192 codes.
// Score = ||c||^2 - 2*(f . c); f.c via hi/lo-split f16 WMMA (near-fp32).
// Software-pipelined: iteration j+1's B loads are issued before iteration j's
// WMMAs so the per-iteration s_wait_loadcnt is covered by matrix work.
// A layout (16-bit 16x32): lane M=L&15, element e -> K = hi*16 + e
// B layout (16-bit 32x16): lane N=L&15, element e -> K = hi*16 + e
// C/D layout: lane N=L&15, vgpr k -> M = k + 8*hi
// ---------------------------------------------------------------------------
__global__ void __launch_bounds__(32)
nn_kernel(const float* __restrict__ cur,
          const _Float16* __restrict__ cbh, const _Float16* __restrict__ cbl,
          const float* __restrict__ cbsq, int* __restrict__ idx_out) {
  const int lane = threadIdx.x;
  const int mrow = lane & 15;
  const int hi   = lane >> 4;
  const int rowbase = blockIdx.x * 16;

  // convert this wave's 16 A-rows once (f32 -> hi/lo f16) into WMMA layout
  v16h a_hi[2], a_lo[2];
  const float* arow = cur + (size_t)(rowbase + mrow) * C_DIM;
  #pragma unroll
  for (int t = 0; t < 2; ++t) {
    #pragma unroll
    for (int g = 0; g < 2; ++g) {
      const float* p = arow + t * 32 + hi * 16 + g * 8;
      float4 v0 = *(const float4*)(p);
      float4 v1 = *(const float4*)(p + 4);
      float vals[8] = {v0.x, v0.y, v0.z, v0.w, v1.x, v1.y, v1.z, v1.w};
      #pragma unroll
      for (int q = 0; q < 8; ++q) {
        _Float16 h = (_Float16)vals[q];
        a_hi[t][g * 8 + q] = h;
        a_lo[t][g * 8 + q] = (_Float16)(vals[q] - (float)h);
      }
    }
  }

  float best[8];
  int   bidx[8];
  #pragma unroll
  for (int k = 0; k < 8; ++k) { best[k] = 3.402823466e38f; bidx[k] = 0; }

  // per-lane base pointers into the pre-split f16 codebook
  const _Float16* ph = cbh + (size_t)mrow * C_DIM + hi * 16;
  const _Float16* pl = cbl + (size_t)mrow * C_DIM + hi * 16;
  const int TILE_STRIDE = 16 * C_DIM;          // halfs per 16-code tile

  // prologue: load tile 0
  v16h bh0 = *(const v16h*)(ph);
  v16h bh1 = *(const v16h*)(ph + 32);
  v16h bl0 = *(const v16h*)(pl);
  v16h bl1 = *(const v16h*)(pl + 32);
  float csq = cbsq[mrow];

  for (int jt = 0; jt < VCODES / 16; ++jt) {
    const int code = jt * 16 + mrow;

    // issue next tile's loads first (double buffer) so the matrix pipe covers
    // the memory latency of this iteration's wait
    v16h nh0 = bh0, nh1 = bh1, nl0 = bl0, nl1 = bl1;
    float ncsq = csq;
    if (jt + 1 < VCODES / 16) {
      const _Float16* qh = ph + (size_t)(jt + 1) * TILE_STRIDE;
      const _Float16* ql = pl + (size_t)(jt + 1) * TILE_STRIDE;
      __builtin_prefetch(qh + TILE_STRIDE, 0, 1);   // global_prefetch_b8, 2 ahead
      nh0 = *(const v16h*)(qh);
      nh1 = *(const v16h*)(qh + 32);
      nl0 = *(const v16h*)(ql);
      nl1 = *(const v16h*)(ql + 32);
      ncsq = cbsq[code + 16];
    }

    v8f acc = {};
    acc = __builtin_amdgcn_wmma_f32_16x16x32_f16(false, a_hi[0], false, bh0,
                                                 (short)0, acc, false, false);
    acc = __builtin_amdgcn_wmma_f32_16x16x32_f16(false, a_lo[0], false, bh0,
                                                 (short)0, acc, false, false);
    acc = __builtin_amdgcn_wmma_f32_16x16x32_f16(false, a_hi[0], false, bl0,
                                                 (short)0, acc, false, false);
    acc = __builtin_amdgcn_wmma_f32_16x16x32_f16(false, a_hi[1], false, bh1,
                                                 (short)0, acc, false, false);
    acc = __builtin_amdgcn_wmma_f32_16x16x32_f16(false, a_lo[1], false, bh1,
                                                 (short)0, acc, false, false);
    acc = __builtin_amdgcn_wmma_f32_16x16x32_f16(false, a_hi[1], false, bl1,
                                                 (short)0, acc, false, false);

    #pragma unroll
    for (int k = 0; k < 8; ++k) {
      float score = csq - 2.0f * acc[k];
      if (score < best[k]) { best[k] = score; bidx[k] = code; }
    }

    bh0 = nh0; bh1 = nh1; bl0 = nl0; bl1 = nl1; csq = ncsq;
  }

  // reduce across the 16 column-class lanes (prefer lowest index on ties)
  #pragma unroll
  for (int k = 0; k < 8; ++k) {
    #pragma unroll
    for (int off = 1; off < 16; off <<= 1) {
      float ob = __shfl_xor(best[k], off, 32);
      int   oi = __shfl_xor(bidx[k], off, 32);
      if (ob < best[k] || (ob == best[k] && oi < bidx[k])) { best[k] = ob; bidx[k] = oi; }
    }
  }
  if (mrow == 0) {
    #pragma unroll
    for (int k = 0; k < 8; ++k)
      idx_out[rowbase + hi * 8 + k] = bidx[k];
  }
}

// ---------------------------------------------------------------------------
// gather codebook + bicubic (Keys a=-0.5) upsample -> emb NHWC [NPIX, 64]
// writes f32 (for blend/recon) and f16 (for WMMA conv A-side)
// ---------------------------------------------------------------------------
__device__ __forceinline__ float keysw(float x) {
  x = fabsf(x);
  if (x <= 1.0f) return (1.5f * x - 2.5f) * x * x + 1.0f;
  if (x <  2.0f) return ((-0.5f * x + 2.5f) * x - 4.0f) * x + 2.0f;
  return 0.0f;
}

__global__ void embed_up_kernel(const int* __restrict__ idx, const float* __restrict__ cb,
                                float* __restrict__ emb, _Float16* __restrict__ embh, int r) {
  int tid = blockIdx.x * blockDim.x + threadIdx.x;
  if (tid >= NTOT) return;
  int c = tid & 63;
  int x = (tid >> 6) & 31;
  int y = (tid >> 11) & 31;
  int b = tid >> 16;

  float acc;
  if (r == HW) {
    int p = (b * HW + y) * HW + x;
    acc = cb[(size_t)idx[p] * C_DIM + c];
  } else {
    float s  = (float)r / (float)HW;
    float py = (y + 0.5f) * s - 0.5f;
    float px = (x + 0.5f) * s - 0.5f;
    int iy = (int)floorf(py), ix = (int)floorf(px);
    float fy = py - (float)iy, fx = px - (float)ix;
    float wy[4], wx[4];
    #pragma unroll
    for (int t = 0; t < 4; ++t) {
      wy[t] = keysw(fy - (float)(t - 1));
      wx[t] = keysw(fx - (float)(t - 1));
    }
    acc = 0.0f;
    #pragma unroll
    for (int ty = 0; ty < 4; ++ty) {
      int cy = min(max(iy - 1 + ty, 0), r - 1);
      #pragma unroll
      for (int tx = 0; tx < 4; ++tx) {
        int cx = min(max(ix - 1 + tx, 0), r - 1);
        int p  = (b * r + cy) * r + cx;
        acc += wy[ty] * wx[tx] * cb[(size_t)idx[p] * C_DIM + c];
      }
    }
  }
  emb[tid]  = acc;
  embh[tid] = (_Float16)acc;
}

// ---------------------------------------------------------------------------
// Phi conv3x3 (implicit-GEMM via WMMA, K = 576) + blend + recon/residual
// update + per-scale SSQ accumulation. One wave = 16 pixels x 16 out-channels.
// A and B operands are pre-converted f16: loop is loads + 18 WMMAs.
// ---------------------------------------------------------------------------
__global__ void __launch_bounds__(32)
conv_blend_kernel(const float* __restrict__ emb,      // NHWC f32 [NPIX,64]
                  const _Float16* __restrict__ embh,  // NHWC f16 [NPIX,64]
                  const _Float16* __restrict__ wrh,   // [64][576] f16
                  const float* __restrict__ bias,     // [64]
                  const float* __restrict__ f,        // NCHW
                  float* __restrict__ recon, float* __restrict__ residual,
                  float* __restrict__ lossAcc, int scale) {
  const int lane = threadIdx.x;
  const int mrow = lane & 15;
  const int hi   = lane >> 4;
  const int pixbase = (int)(blockIdx.x >> 2) * 16;
  const int otile   = (int)(blockIdx.x & 3);

  // pixel this lane supplies for the A matrix
  const int apx = pixbase + mrow;
  const int ax = apx & 31, ay = (apx >> 5) & 31, ab = apx >> 10;
  const int o  = otile * 16 + mrow;            // B column this lane supplies
  const _Float16* browh = wrh + (size_t)o * KCONV;

  v8f acc = {};
  #pragma unroll
  for (int t = 0; t < KCONV / 32; ++t) {       // 18 k-tiles of 32
    v8h ag[2];
    #pragma unroll
    for (int g = 0; g < 2; ++g) {
      int K0  = t * 32 + hi * 16 + g * 8;      // 8 consecutive K within one tap
      int tap = K0 >> 6, c0 = K0 & 63;
      int ky = tap / 3, kx = tap % 3;
      int yy = ay + ky - 1, xx = ax + kx - 1;
      v8h a = {};
      if (yy >= 0 && yy < HW && xx >= 0 && xx < HW)
        a = *(const v8h*)(embh + (size_t)((ab * HW + yy) * HW + xx) * C_DIM + c0);
      ag[g] = a;
    }
    v16h Am = __builtin_shufflevector(ag[0], ag[1], 0, 1, 2, 3, 4, 5, 6, 7,
                                      8, 9, 10, 11, 12, 13, 14, 15);
    v16h Bm = *(const v16h*)(browh + t * 32 + hi * 16);
    acc = __builtin_amdgcn_wmma_f32_16x16x32_f16(false, Am, false, Bm,
                                                 (short)0, acc, false, false);
  }

  const float bo = bias[o];
  float lsum = 0.0f;
  #pragma unroll
  for (int k = 0; k < 8; ++k) {
    int p = pixbase + hi * 8 + k;              // output row M = k + 8*hi
    int x = p & 31, y = (p >> 5) & 31, b = p >> 10;
    float convres = acc[k] + bo;
    float ein = emb[(size_t)p * C_DIM + o];
    float e = (1.0f - RATIO_C) * ein + RATIO_C * convres;
    size_t nchw = (((size_t)b * C_DIM + o) * HW + y) * HW + x;
    float rc = recon[nchw] + e;
    recon[nchw] = rc;
    residual[nchw] -= e;
    float d = rc - f[nchw];
    lsum += d * d;
  }
  #pragma unroll
  for (int off = 16; off >= 1; off >>= 1) lsum += __shfl_xor(lsum, off, 32);
  if (lane == 0) atomicAdd(&lossAcc[scale], lsum);
}

// ---------------------------------------------------------------------------
// out[0..N) = (recon - f) + f ; out[N] = total loss
// ---------------------------------------------------------------------------
__global__ void final_kernel(const float* __restrict__ recon, const float* __restrict__ f,
                             const float* __restrict__ lossAcc, float* __restrict__ out) {
  int i = blockIdx.x * blockDim.x + threadIdx.x;
  if (i < NTOT) out[i] = (recon[i] - f[i]) + f[i];
  if (i == 0) {
    float tl = 0.0f;
    for (int s = 0; s < NSCALES; ++s)
      tl += (1.0f + COMMIT_C) * (lossAcc[s] / (float)NTOT);
    out[NTOT] = tl / (float)NSCALES;
  }
}

// ---------------------------------------------------------------------------
extern "C" void kernel_launch(void* const* d_in, const int* in_sizes, int n_in,
                              void* d_out, int out_size, void* d_ws, size_t ws_size,
                              hipStream_t stream) {
  const float* f     = (const float*)d_in[0];
  const float* cb    = (const float*)d_in[1];
  const float* phi_w = (const float*)d_in[2];
  const float* phi_b = (const float*)d_in[3];
  float* out = (float*)d_out;

  char* ws = (char*)d_ws;
  size_t off = 0;
  auto wsalloc = [&](size_t bytes) -> void* {
    void* p = ws + off;
    off += (bytes + 255) & ~(size_t)255;
    return p;
  };
  float*    residual = (float*)   wsalloc((size_t)NTOT * 4);
  float*    recon    = (float*)   wsalloc((size_t)NTOT * 4);
  float*    buf      = (float*)   wsalloc((size_t)NTOT * 4);  // cur / emb (NHWC f32)
  _Float16* embh     = (_Float16*)wsalloc((size_t)NTOT * 2);  // emb (NHWC f16)
  _Float16* cbh      = (_Float16*)wsalloc((size_t)VCODES * C_DIM * 2);
  _Float16* cbl      = (_Float16*)wsalloc((size_t)VCODES * C_DIM * 2);
  int*      idx      = (int*)     wsalloc((size_t)NPIX * 4);
  float*    cbsq     = (float*)   wsalloc((size_t)VCODES * 4);
  _Float16* wrh      = (_Float16*)wsalloc((size_t)C_DIM * KCONV * 2);
  float*    lossAcc  = (float*)   wsalloc((size_t)NSCALES * 4);

  init_kernel<<<(NTOT + 255) / 256, 256, 0, stream>>>(f, residual, recon, lossAcc);
  cbsq_kernel<<<(VCODES + 255) / 256, 256, 0, stream>>>(cb, cbsq);
  cvt_cb_kernel<<<(VCODES * C_DIM + 255) / 256, 256, 0, stream>>>(cb, cbh, cbl);

  const int RESL[NSCALES] = {1, 2, 4, 8, 16, 32};
  for (int i = 0; i < NSCALES; ++i) {
    const int r  = RESL[i];
    const int Nr = BATCH * r * r;             // rows for NN search (multiple of 16)
    repack_w_kernel<<<(C_DIM * KCONV + 255) / 256, 256, 0, stream>>>(phi_w, wrh, i);
    const int ncur = Nr * C_DIM;
    down_kernel<<<(ncur + 255) / 256, 256, 0, stream>>>(residual, buf, r);
    nn_kernel<<<Nr / 16, 32, 0, stream>>>(buf, cbh, cbl, cbsq, idx);
    embed_up_kernel<<<(NTOT + 255) / 256, 256, 0, stream>>>(idx, cb, buf, embh, r);
    conv_blend_kernel<<<(NPIX / 16) * 4, 32, 0, stream>>>(buf, embh, wrh,
                                                          phi_b + (size_t)i * C_DIM,
                                                          f, recon, residual, lossAcc, i);
  }
  final_kernel<<<(NTOT + 255) / 256, 256, 0, stream>>>(recon, f, lossAcc, out);
}
